// MultiHeadAttention_48498770707023
// MI455X (gfx1250) — compile-verified, matrix operand order checked by reading, not run on hardware
//
#include <hip/hip_runtime.h>

typedef __attribute__((ext_vector_type(16))) __bf16 v16bf;
typedef __attribute__((ext_vector_type(8)))  float  v8f;
typedef unsigned short u16b;

#define DIM   1024
#define NH    16
#define HD    64
#define BB    4
#define TT    2048
#define NTOK  (BB*TT)

// ---------- helpers ----------

__device__ __forceinline__ u16b f2bf(float f) {
  union { float f; unsigned u; } v; v.f = f;
  unsigned r = v.u + 0x7FFFu + ((v.u >> 16) & 1u);   // round-to-nearest-even
  return (u16b)(r >> 16);
}

// Load one 16x32 bf16 A/B^T fragment (ISA 16-bit A layout):
// lanes 0-15: row=lane, K chunks [0..7],[16..23]; lanes 16-31: row=lane-16, chunks [8..15],[24..31]
__device__ __forceinline__ v16bf load_frag(const u16b* __restrict__ base, int ld, int lane) {
  const int hi = lane >> 4, lo = lane & 15;
  const u16b* p = base + (size_t)lo * ld + hi * 8;
  union { v16bf v; uint4 u[2]; } f;
  f.u[0] = *(const uint4*)(p);
  f.u[1] = *(const uint4*)(p + 16);
  return f.v;
}

__device__ __forceinline__ v8f wmma_bf16(v16bf a, v16bf b, v8f c) {
  return __builtin_amdgcn_wmma_f32_16x16x32_bf16(false, a, false, b, (short)0, c, false, false);
}

// CDNA5 async copy: 16B global -> LDS, tracked with ASYNCcnt.
// Generic LDS address truncates to the LDS byte offset (ISA 10.2).
__device__ __forceinline__ void async_cp16(u16b* lds_dst, const u16b* g_src) {
  unsigned loff = (unsigned)(unsigned long long)lds_dst;
  asm volatile("global_load_async_to_lds_b128 %0, %1, off"
               :: "v"(loff), "v"(g_src) : "memory");
}

// ---------- stage 0: casts / transposes ----------

__global__ __launch_bounds__(256) void k_cast_bf16(const float* __restrict__ in,
                                                   u16b* __restrict__ out, int n) {
  int i = blockIdx.x * 256 + threadIdx.x;
  if (i < n) out[i] = f2bf(in[i]);
}

// in: K x N (row-major), out: N x K (row-major) = B^T, cast to bf16
__global__ __launch_bounds__(256) void k_transpose_cast(const float* __restrict__ in,
                                                        u16b* __restrict__ out, int K, int N) {
  int i = blockIdx.x * 256 + threadIdx.x;
  if (i < K * N) {
    int n = i / K, k = i - n * K;
    out[i] = f2bf(in[(size_t)k * N + n]);
  }
}

// ---------- stage 1: QKV projection GEMM (async-LDS double buffered) ----------
// A: x_bf16 [8192 x 1024], Bt: Wqkv^T [3072 x 1024], bias [3072]
// writes Q,K as (B,H,T,D) bf16 and V^T as (B,H,D,T) bf16
__global__ __launch_bounds__(256) void k_gemm_qkv(const u16b* __restrict__ A,
                                                  const u16b* __restrict__ Bt,
                                                  const float* __restrict__ bias,
                                                  u16b* __restrict__ Q,
                                                  u16b* __restrict__ Kc,
                                                  u16b* __restrict__ Vt) {
  __shared__ __align__(16) u16b shA[2][128 * 64];   // 2 x 16KB
  __shared__ __align__(16) u16b shB[2][64 * 64];    // 2 x 8KB
  const int tid = threadIdx.x;
  const int lane = tid & 31, w = tid >> 5;
  const int wm = w & 3, wn = w >> 2;
  const int m0b = blockIdx.y * 128;
  const int n0b = blockIdx.x * 64;
  v8f acc[2][2] = {};

  auto issue = [&](int buf, int kt) {
    const u16b* Ab = A  + (size_t)m0b * DIM + kt * 64;
    const u16b* Bb = Bt + (size_t)n0b * DIM + kt * 64;
#pragma unroll
    for (int i = 0; i < 4; i++) {                 // A tile: 128 rows x 64 cols = 1024 16B chunks
      int c = i * 256 + tid;
      int row = c >> 3, ch = c & 7;
      async_cp16(&shA[buf][c * 8], Ab + (size_t)row * DIM + ch * 8);
    }
#pragma unroll
    for (int i = 0; i < 2; i++) {                 // B tile: 64 rows x 64 cols = 512 chunks
      int c = i * 256 + tid;
      int row = c >> 3, ch = c & 7;
      async_cp16(&shB[buf][c * 8], Bb + (size_t)row * DIM + ch * 8);
    }
  };

  issue(0, 0);
  for (int kt = 0; kt < DIM / 64; kt++) {
    const int buf = kt & 1;
    if (kt + 1 < DIM / 64) {
      issue(buf ^ 1, kt + 1);
      asm volatile("s_wait_asynccnt 0x6" ::: "memory");   // tile kt landed (6 newer pending)
    } else {
      asm volatile("s_wait_asynccnt 0x0" ::: "memory");
    }
    __syncthreads();
    const u16b* At = &shA[buf][(wm * 32) * 64];
    const u16b* Bw = &shB[buf][(wn * 32) * 64];
#pragma unroll
    for (int kk = 0; kk < 64; kk += 32) {
      v16bf a0 = load_frag(At + kk, 64, lane);
      v16bf a1 = load_frag(At + 16 * 64 + kk, 64, lane);
      v16bf b0 = load_frag(Bw + kk, 64, lane);
      v16bf b1 = load_frag(Bw + 16 * 64 + kk, 64, lane);
      acc[0][0] = wmma_bf16(a0, b0, acc[0][0]);
      acc[0][1] = wmma_bf16(a0, b1, acc[0][1]);
      acc[1][0] = wmma_bf16(a1, b0, acc[1][0]);
      acc[1][1] = wmma_bf16(a1, b1, acc[1][1]);
    }
    __syncthreads();                                       // WAR on recycled buffer
  }

  const int hi = lane >> 4, lo = lane & 15;
  const int m0 = m0b + wm * 32, n0 = n0b + wn * 32;
  for (int tm = 0; tm < 2; tm++)
    for (int tn = 0; tn < 2; tn++)
#pragma unroll
      for (int i = 0; i < 8; i++) {
        int row = m0 + tm * 16 + hi * 8 + i;
        int col = n0 + tn * 16 + lo;
        float val = acc[tm][tn][i] + bias[col];
        u16b bv = f2bf(val);
        int b = row >> 11, t = row & (TT - 1);
        int which = col >> 10, c = col & (DIM - 1);
        int h = c >> 6, d = c & 63;
        size_t bh = (size_t)b * NH + h;
        if (which == 0)      Q [(bh * TT + t) * HD + d] = bv;
        else if (which == 1) Kc[(bh * TT + t) * HD + d] = bv;
        else                 Vt[(bh * HD + d) * TT + t] = bv;
      }
}

// ---------- stage 2: causal flash attention ----------
// Q,K: (B,H,T,64) bf16 ; Vt: (B,H,64,T) bf16 ; O: (B,T,H,64) bf16
__global__ __launch_bounds__(128) void k_attn(const u16b* __restrict__ Q,
                                              const u16b* __restrict__ K,
                                              const u16b* __restrict__ Vt,
                                              u16b* __restrict__ O) {
  __shared__ __align__(16) u16b plds[4][16 * 32];   // per-wave P tile (16 rows x 32 kv)
  const int lane = threadIdx.x & 31, w = threadIdx.x >> 5;
  const int hi = lane >> 4, lo = lane & 15;
  const int bh = blockIdx.y;               // 0..63
  const int b = bh >> 4, h = bh & 15;
  const int qbase = blockIdx.x * 64 + w * 16;
  const u16b* Qb = Q  + (size_t)bh * TT * HD;
  const u16b* Kb = K  + (size_t)bh * TT * HD;
  const u16b* Vb = Vt + (size_t)bh * HD * TT;

  v16bf q0 = load_frag(Qb + (size_t)qbase * HD +  0, HD, lane);
  v16bf q1 = load_frag(Qb + (size_t)qbase * HD + 32, HD, lane);

  v8f o[4] = {};                 // d-columns t*16 .. t*16+15
  float m_i[8], l_i[8];
#pragma unroll
  for (int i = 0; i < 8; i++) { m_i[i] = -1e30f; l_i[i] = 0.f; }
  const float scale = 0.125f;    // 1/sqrt(64)

  const int jend = (qbase + 15) >> 5;      // last 32-wide kv block touching causal region
  for (int jb = 0; jb <= jend; jb++) {
    const int j0 = jb * 32;
    if (j0 + 32 < TT) {                                   // prefetch next KV tile
      __builtin_prefetch(Kb + (size_t)(j0 + 32) * HD, 0, 0);
      __builtin_prefetch(Vb + (size_t)0 * TT + j0 + 32, 0, 0);
    }
    v8f s0 = {}, s1 = {};
    {
      v16bf kb;
      kb = load_frag(Kb + (size_t)j0 * HD +  0, HD, lane);        s0 = wmma_bf16(q0, kb, s0);
      kb = load_frag(Kb + (size_t)j0 * HD + 32, HD, lane);        s0 = wmma_bf16(q1, kb, s0);
      kb = load_frag(Kb + (size_t)(j0 + 16) * HD +  0, HD, lane); s1 = wmma_bf16(q0, kb, s1);
      kb = load_frag(Kb + (size_t)(j0 + 16) * HD + 32, HD, lane); s1 = wmma_bf16(q1, kb, s1);
    }
#pragma unroll
    for (int i = 0; i < 8; i++) {
      const int qg = qbase + hi * 8 + i;
      float x0 = (j0 + lo      <= qg) ? s0[i] * scale : -1e30f;
      float x1 = (j0 + 16 + lo <= qg) ? s1[i] * scale : -1e30f;
      float mx = fmaxf(x0, x1);
#pragma unroll
      for (int off = 8; off; off >>= 1) mx = fmaxf(mx, __shfl_xor(mx, off, 16));
      const float mnew  = fmaxf(m_i[i], mx);
      const float alpha = __expf(m_i[i] - mnew);
      const float p0 = __expf(x0 - mnew);
      const float p1 = __expf(x1 - mnew);
      float rs = p0 + p1;
#pragma unroll
      for (int off = 8; off; off >>= 1) rs += __shfl_xor(rs, off, 16);
      l_i[i] = l_i[i] * alpha + rs;
      m_i[i] = mnew;
#pragma unroll
      for (int t = 0; t < 4; t++) o[t][i] *= alpha;
      const int row = hi * 8 + i;
      plds[w][row * 32 + lo]      = f2bf(p0);
      plds[w][row * 32 + 16 + lo] = f2bf(p1);
    }
    asm volatile("s_wait_dscnt 0x0" ::: "memory");   // LDS RAW within wave
    // reload P as an A-fragment (16x32)
    union { v16bf v; uint4 u[2]; } pf;
    const uint4* pl = (const uint4*)&plds[w][lo * 32];
    pf.u[0] = pl[hi ? 1 : 0];
    pf.u[1] = pl[hi ? 3 : 2];
#pragma unroll
    for (int t = 0; t < 4; t++) {
      v16bf vb = load_frag(Vb + (size_t)(t * 16) * TT + j0, TT, lane);
      o[t] = wmma_bf16(pf.v, vb, o[t]);
    }
  }
  // epilogue: normalize and store (B,T,H,D) bf16
#pragma unroll
  for (int t = 0; t < 4; t++)
#pragma unroll
    for (int i = 0; i < 8; i++) {
      const int row = qbase + hi * 8 + i;
      const float val = o[t][i] / l_i[i];
      const size_t tok = (size_t)b * TT + row;
      O[tok * DIM + h * HD + t * 16 + lo] = f2bf(val);
    }
}

// ---------- stage 3: output projection GEMM (async-LDS double buffered) ----------
// A: attn_out bf16 [8192 x 1024], Bt: Wout^T [1024 x 1024], bias [1024], C fp32 [8192 x 1024]
__global__ __launch_bounds__(256) void k_gemm_out(const u16b* __restrict__ A,
                                                  const u16b* __restrict__ Bt,
                                                  const float* __restrict__ bias,
                                                  float* __restrict__ C) {
  __shared__ __align__(16) u16b shA[2][128 * 64];
  __shared__ __align__(16) u16b shB[2][64 * 64];
  const int tid = threadIdx.x;
  const int lane = tid & 31, w = tid >> 5;
  const int wm = w & 3, wn = w >> 2;
  const int m0b = blockIdx.y * 128;
  const int n0b = blockIdx.x * 64;
  v8f acc[2][2] = {};

  auto issue = [&](int buf, int kt) {
    const u16b* Ab = A  + (size_t)m0b * DIM + kt * 64;
    const u16b* Bb = Bt + (size_t)n0b * DIM + kt * 64;
#pragma unroll
    for (int i = 0; i < 4; i++) {
      int c = i * 256 + tid;
      int row = c >> 3, ch = c & 7;
      async_cp16(&shA[buf][c * 8], Ab + (size_t)row * DIM + ch * 8);
    }
#pragma unroll
    for (int i = 0; i < 2; i++) {
      int c = i * 256 + tid;
      int row = c >> 3, ch = c & 7;
      async_cp16(&shB[buf][c * 8], Bb + (size_t)row * DIM + ch * 8);
    }
  };

  issue(0, 0);
  for (int kt = 0; kt < DIM / 64; kt++) {
    const int buf = kt & 1;
    if (kt + 1 < DIM / 64) {
      issue(buf ^ 1, kt + 1);
      asm volatile("s_wait_asynccnt 0x6" ::: "memory");
    } else {
      asm volatile("s_wait_asynccnt 0x0" ::: "memory");
    }
    __syncthreads();
    const u16b* At = &shA[buf][(wm * 32) * 64];
    const u16b* Bw = &shB[buf][(wn * 32) * 64];
#pragma unroll
    for (int kk = 0; kk < 64; kk += 32) {
      v16bf a0 = load_frag(At + kk, 64, lane);
      v16bf a1 = load_frag(At + 16 * 64 + kk, 64, lane);
      v16bf b0 = load_frag(Bw + kk, 64, lane);
      v16bf b1 = load_frag(Bw + 16 * 64 + kk, 64, lane);
      acc[0][0] = wmma_bf16(a0, b0, acc[0][0]);
      acc[0][1] = wmma_bf16(a0, b1, acc[0][1]);
      acc[1][0] = wmma_bf16(a1, b0, acc[1][0]);
      acc[1][1] = wmma_bf16(a1, b1, acc[1][1]);
    }
    __syncthreads();
  }

  const int hi = lane >> 4, lo = lane & 15;
  const int m0 = m0b + wm * 32, n0 = n0b + wn * 32;
  for (int tm = 0; tm < 2; tm++)
    for (int tn = 0; tn < 2; tn++)
#pragma unroll
      for (int i = 0; i < 8; i++) {
        int row = m0 + tm * 16 + hi * 8 + i;
        int col = n0 + tn * 16 + lo;
        C[(size_t)row * DIM + col] = acc[tm][tn][i] + bias[col];
      }
}

// ---------- host launch ----------

extern "C" void kernel_launch(void* const* d_in, const int* in_sizes, int n_in,
                              void* d_out, int out_size, void* d_ws, size_t ws_size,
                              hipStream_t stream) {
  const float* x     = (const float*)d_in[0];  // (4,2048,1024)
  const float* Wqkv  = (const float*)d_in[1];  // (1024,3072)
  const float* bqkv  = (const float*)d_in[2];  // (3072)
  const float* Wout  = (const float*)d_in[3];  // (1024,1024)
  const float* bout  = (const float*)d_in[4];  // (1024)
  float* out = (float*)d_out;

  char* ws = (char*)d_ws;
  size_t off = 0;
  u16b* xbf   = (u16b*)(ws + off); off += (size_t)NTOK * DIM * 2;        // 16 MiB
  u16b* wqkvT = (u16b*)(ws + off); off += (size_t)3 * DIM * DIM * 2;     //  6 MiB
  u16b* woutT = (u16b*)(ws + off); off += (size_t)DIM * DIM * 2;         //  2 MiB
  u16b* Qb    = (u16b*)(ws + off); off += (size_t)NTOK * DIM * 2;        // 16 MiB
  u16b* Kb    = (u16b*)(ws + off); off += (size_t)NTOK * DIM * 2;        // 16 MiB
  u16b* Vt    = (u16b*)(ws + off); off += (size_t)NTOK * DIM * 2;        // 16 MiB
  u16b* AO    = (u16b*)(ws + off); off += (size_t)NTOK * DIM * 2;        // 16 MiB

  // stage 0: casts
  {
    int n = NTOK * DIM;
    k_cast_bf16<<<(n + 255) / 256, 256, 0, stream>>>(x, xbf, n);
    int nq = 3 * DIM * DIM;
    k_transpose_cast<<<(nq + 255) / 256, 256, 0, stream>>>(Wqkv, wqkvT, DIM, 3 * DIM);
    int no = DIM * DIM;
    k_transpose_cast<<<(no + 255) / 256, 256, 0, stream>>>(Wout, woutT, DIM, DIM);
  }
  // stage 1: QKV projection, 8192x3072x1024
  k_gemm_qkv<<<dim3(3 * DIM / 64, NTOK / 128), 256, 0, stream>>>(xbf, wqkvT, bqkv, Qb, Kb, Vt);
  // stage 2: flash attention
  k_attn<<<dim3(TT / 64, BB * NH), 128, 0, stream>>>(Qb, Kb, Vt, AO);
  // stage 3: output projection, 8192x1024x1024
  k_gemm_out<<<dim3(DIM / 64, NTOK / 128), 256, 0, stream>>>(AO, woutT, bout, out);
}
